// LSTMPredictor_3659312136610
// MI455X (gfx1250) — matrix operand order repaired and checked by reading, not compiled, not run on hardware
//
#include <hip/hip_runtime.h>

#define T_STEPS 1024
#define F_IN 5
#define H_REAL 51
#define WAVES_PER_BLOCK 4
#define THREADS (WAVES_PER_BLOCK * 32)

typedef unsigned int uint;
typedef unsigned short ushort;
typedef __attribute__((ext_vector_type(16))) __bf16 v16bf;
typedef __attribute__((ext_vector_type(2)))  __bf16 v2bf;
typedef __attribute__((ext_vector_type(8)))  float  v8f;
typedef __attribute__((ext_vector_type(4)))  uint   v4u;
typedef __attribute__((ext_vector_type(2)))  uint   v2u;

union Frag { uint u[8]; v16bf v; v4u q[2]; };

__device__ __forceinline__ uint bf16r(float a) {
  uint u = __float_as_uint(a);
  u += 0x7FFFu + ((u >> 16) & 1u);   // round-to-nearest-even
  return u >> 16;
}

#if __has_builtin(__builtin_amdgcn_cvt_pk_bf16_f32)
__device__ __forceinline__ uint pack2(float a, float b) {
  union { v2bf v; uint u; } cv;
  cv.v = __builtin_amdgcn_cvt_pk_bf16_f32(a, b);
  return cv.u;
}
#elif __has_builtin(__builtin_amdgcn_perm)
__device__ __forceinline__ uint pack2(float a, float b) {
  uint ua = __float_as_uint(a);
  uint ub = __float_as_uint(b);
  ua += 0x7FFFu + ((ua >> 16) & 1u);                 // v_add3 (RNE bias)
  ub += 0x7FFFu + ((ub >> 16) & 1u);                 // v_add3
  return __builtin_amdgcn_perm(ub, ua, 0x07060302u); // {ub.hi, ua.hi}
}
#else
__device__ __forceinline__ uint pack2(float a, float b) {
  return bf16r(a) | (bf16r(b) << 16);
}
#endif

__device__ __forceinline__ float t_tanh(float x) {
#if __has_builtin(__builtin_amdgcn_tanh_f32)
  return __builtin_amdgcn_tanh_f32(x);
#elif __has_builtin(__builtin_amdgcn_tanhf)
  return __builtin_amdgcn_tanhf(x);
#else
  return 1.0f - 2.0f / (1.0f + __expf(2.0f * x)); // saturates correctly at +/-1
#endif
}
// sigmoid(x) where the accumulator already holds x/2 (weights pre-scaled by 0.5)
__device__ __forceinline__ float t_sig_pre(float y) {
  return fmaf(0.5f, t_tanh(y), 0.5f);
}

__device__ __forceinline__ v8f wmma_bf16(v16bf a, v16bf b, v8f c) {
  // (neg_a, A, neg_b, B, c_mod, C, reuse_a, reuse_b)
  return __builtin_amdgcn_wmma_f32_16x16x32_bf16(false, a, false, b, (short)0, c, false, false);
}

// A-fragment (16x32 bf16) from LDS: lanes 0-15 take K {0..15}, lanes 16-31 K {16..31} (this code's order; B matches)
__device__ __forceinline__ v16bf load_afrag(const ushort* wp, int ko) {
  Frag a;
  a.q[0] = *(const v4u*)(wp + ko);        // K ko..ko+7
  a.q[1] = *(const v4u*)(wp + ko + 8);    // K ko+8..ko+15
  return a.v;
}

// Convert two C-layout row groups into one B-fragment (K=32).
// Pack pairs first, then exchange packed words between the two lane halves:
//   lane<16 : u[k] = own P0[k]        u[4+k] = partner P0[k]
//   lane>=16: u[k] = partner P1[k]    u[4+k] = own P1[k]
__device__ __forceinline__ void build_frag(const float S0[8], const float S1[8],
                                           bool lo, Frag& f) {
#pragma unroll
  for (int k = 0; k < 4; ++k) {
    uint p0 = pack2(S0[2*k], S0[2*k+1]);
    uint p1 = pack2(S1[2*k], S1[2*k+1]);
#if __has_builtin(__builtin_amdgcn_permlane16_swap)
    // a' = {p0[0:15], p1[0:15]->hi}, b' = {p0[16:31]->lo, p1[16:31]}
    v2u r = __builtin_amdgcn_permlane16_swap(p0, p1, false, false);
    f.u[k]     = r.x;
    f.u[4 + k] = r.y;
#else
    uint q0 = __shfl_xor(p0, 16, 32);
    uint q1 = __shfl_xor(p1, 16, 32);
    f.u[k]     = lo ? p0 : q1;
    f.u[4 + k] = lo ? q0 : p1;
#endif
  }
}

__global__ __launch_bounds__(THREADS)
__attribute__((amdgpu_num_vgpr(256)))
void lstm_wmma_kernel(const float* __restrict__ x,
                      const float* __restrict__ W_ih1, const float* __restrict__ W_hh1,
                      const float* __restrict__ b_ih1, const float* __restrict__ b_hh1,
                      const float* __restrict__ W_ih2, const float* __restrict__ W_hh2,
                      const float* __restrict__ b_ih2, const float* __restrict__ b_hh2,
                      const float* __restrict__ W_lin, const float* __restrict__ b_lin,
                      float* __restrict__ out) {
  // Padded transposed weights in LDS, bf16, WMMA-A row-major [gate-row 0..255][K].
  // Gate rows for i/f/o (blocks 0,1,3) are pre-scaled by 0.5 so sigmoid needs no input mul.
  __shared__ ushort sW1[256 * 64];    // layer1: K=64  (h1[0..50], bias row 51, x rows 52..56)
  __shared__ ushort sW2[256 * 128];   // layer2: K=128 (h1 block 0..63 w/ bias @51, h2 block 64..127)
  __shared__ __align__(32) float sWlin[64];

  const int tid = threadIdx.x;

  for (int idx = tid; idx < 256 * 64; idx += THREADS) {
    int r = idx >> 6, k = idx & 63;
    int g = r >> 6, p = r & 63;
    float v = 0.f;
    if (p < H_REAL) {
      int sr = g * H_REAL + p;
      if (k < H_REAL)            v = W_hh1[sr * H_REAL + k];
      else if (k == H_REAL)      v = b_ih1[sr] + b_hh1[sr];
      else if (k < H_REAL + 1 + F_IN) v = W_ih1[sr * F_IN + (k - (H_REAL + 1))];
    }
    if (g != 2) v *= 0.5f;   // pre-scale sigmoid gates (exact in bf16)
    sW1[idx] = (ushort)bf16r(v);
  }
  for (int idx = tid; idx < 256 * 128; idx += THREADS) {
    int r = idx >> 7, k = idx & 127;
    int g = r >> 6, p = r & 63;
    float v = 0.f;
    if (p < H_REAL) {
      int sr = g * H_REAL + p;
      if (k < 64) {
        if (k < H_REAL)       v = W_ih2[sr * H_REAL + k];
        else if (k == H_REAL) v = b_ih2[sr] + b_hh2[sr];
      } else {
        int q = k - 64;
        if (q < H_REAL)       v = W_hh2[sr * H_REAL + q];
      }
    }
    if (g != 2) v *= 0.5f;   // pre-scale sigmoid gates (exact in bf16)
    sW2[idx] = (ushort)bf16r(v);
  }
  for (int idx = tid; idx < 64; idx += THREADS)
    sWlin[idx] = (idx < H_REAL) ? W_lin[idx] : 0.f;
  __syncthreads();

  const int  lane   = tid & 31;
  const int  wave   = tid >> 5;
  const bool hiHalf = lane >= 16;
  const bool loHalf = !hiHalf;
  const int  laneLo = lane & 15;
  const int  b0     = (blockIdx.x * WAVES_PER_BLOCK + wave) * 16;
  const int  ko     = hiHalf ? 16 : 0;

  const float blin = b_lin[0];

  float c1r[32], c2r[32];
#pragma unroll
  for (int i = 0; i < 32; ++i) { c1r[i] = 0.f; c2r[i] = 0.f; }

  Frag h1f0, h1f1, h2f0, h2f1;
#pragma unroll
  for (int i = 0; i < 8; ++i) { h1f0.u[i] = 0; h1f1.u[i] = 0; h2f0.u[i] = 0; h2f1.u[i] = 0; }
  if (hiHalf) h1f1.u[1] = 0x3F800000u;  // K-row 51 = 1.0 (bias row), row 50 = 0

  // all lanes load x for their laneLo batch row (hi/lo partners hit identical
  // addresses -> fully coalesced; only hi-lane values are consumed)
  const float* xrow = x + (size_t)(b0 + laneLo) * T_STEPS * F_IN;
  float xv0 = xrow[0], xv1 = xrow[1], xv2 = xrow[2], xv3 = xrow[3], xv4 = xrow[4];

  uint lv = 0;  // opaque zero: defeats LICM on the weight loads without changing addresses

  for (int t = 0; t < T_STEPS; ++t) {
    asm volatile("" : "+v"(lv));   // redefine per iteration -> LDS loads stay in the loop
    const ushort* sW1v = sW1 + lv;
    const ushort* sW2v = sW2 + lv;
    const float*  sWLv = sWlin + lv;

    // fold x_t into h1 fragment K-rows 52..56 (held by hi lanes) -- branchless
    h1f1.u[2] = hiHalf ? pack2(xv0, xv1) : h1f1.u[2];
    h1f1.u[3] = hiHalf ? pack2(xv2, xv3) : h1f1.u[3];
    h1f1.u[4] = hiHalf ? pack2(xv4, 0.0f) : h1f1.u[4];   // row 57 stays 0

    // prefetch x_{t+1} (clamped index, unconditional)
    {
      int tn = t + 1;
      tn = tn < T_STEPS ? tn : (T_STEPS - 1);
      const float* xp = xrow + (size_t)tn * F_IN;
      xv0 = xp[0]; xv1 = xp[1]; xv2 = xp[2]; xv3 = xp[3]; xv4 = xp[4];
    }

    // ---------------- layer 1: G = W1'[256x64] x H1[64x16] ----------------
    {
      float hn[4][8];
#pragma unroll
      for (int jt = 0; jt < 4; ++jt) {
        v8f acc[4];
#pragma unroll
        for (int g = 0; g < 4; ++g) {
          const ushort* wp = &sW1v[((g * 4 + jt) * 16 + laneLo) * 64];
          v16bf a0 = load_afrag(wp, ko);
          v16bf a1 = load_afrag(wp + 32, ko);
          v8f a = {};
          a = wmma_bf16(a0, h1f0.v, a);
          a = wmma_bf16(a1, h1f1.v, a);
          acc[g] = a;
        }
#pragma unroll
        for (int k = 0; k < 8; ++k) {
          float iv = t_sig_pre(acc[0][k]);
          float fv = t_sig_pre(acc[1][k]);
          float gv = t_tanh(acc[2][k]);
          float ov = t_sig_pre(acc[3][k]);
          float c  = fmaf(fv, c1r[jt * 8 + k], iv * gv);
          c1r[jt * 8 + k] = c;
          hn[jt][k] = ov * t_tanh(c);
        }
      }
      // old fragments fully consumed -> build new ones in place (no temp copies)
      build_frag(hn[0], hn[1], loHalf, h1f0);
      build_frag(hn[2], hn[3], loHalf, h1f1);
    }
    // re-arm bias row 51 = 1.0 (hi lanes), branchless
    h1f1.u[1] = hiHalf ? ((h1f1.u[1] & 0x0000FFFFu) | 0x3F800000u) : h1f1.u[1];

    // ---------------- layer 2: G = W2'[256x128] x [H1;H2] ----------------
    float partial = 0.f;
    {
      float hn[4][8];
#pragma unroll
      for (int jt = 0; jt < 4; ++jt) {
        v8f acc[4];
#pragma unroll
        for (int g = 0; g < 4; ++g) {
          const ushort* wp = &sW2v[((g * 4 + jt) * 16 + laneLo) * 128];
          v16bf a0 = load_afrag(wp, ko);
          v16bf a1 = load_afrag(wp + 32, ko);
          v16bf a2 = load_afrag(wp + 64, ko);
          v16bf a3 = load_afrag(wp + 96, ko);
          v8f a = {};
          a = wmma_bf16(a0, h1f0.v, a);
          a = wmma_bf16(a1, h1f1.v, a);
          a = wmma_bf16(a2, h2f0.v, a);
          a = wmma_bf16(a3, h2f1.v, a);
          acc[g] = a;
        }
        // linear-head weights for this row-tile (one 32B LDS vector load)
        const v8f wv = *(const v8f*)&sWLv[jt * 16 + (hiHalf ? 8 : 0)];
#pragma unroll
        for (int k = 0; k < 8; ++k) {
          float iv = t_sig_pre(acc[0][k]);
          float fv = t_sig_pre(acc[1][k]);
          float gv = t_tanh(acc[2][k]);
          float ov = t_sig_pre(acc[3][k]);
          float c  = fmaf(fv, c2r[jt * 8 + k], iv * gv);
          c2r[jt * 8 + k] = c;
          float h = ov * t_tanh(c);
          hn[jt][k] = h;
          partial = fmaf(wv[k], h, partial);
        }
      }
      build_frag(hn[0], hn[1], loHalf, h2f0);
      build_frag(hn[2], hn[3], loHalf, h2f1);
    }

    // linear head: reduce the two lane-halves, relu, store
    partial += __shfl_xor(partial, 16, 32);
    if (loHalf)
      out[(size_t)(b0 + laneLo) * T_STEPS + t] = fmaxf(partial + blin, 0.0f);
  }
}

extern "C" void kernel_launch(void* const* d_in, const int* in_sizes, int n_in,
                              void* d_out, int out_size, void* d_ws, size_t ws_size,
                              hipStream_t stream) {
  const float* x     = (const float*)d_in[0];
  const float* W_ih1 = (const float*)d_in[1];
  const float* W_hh1 = (const float*)d_in[2];
  const float* b_ih1 = (const float*)d_in[3];
  const float* b_hh1 = (const float*)d_in[4];
  const float* W_ih2 = (const float*)d_in[5];
  const float* W_hh2 = (const float*)d_in[6];
  const float* b_ih2 = (const float*)d_in[7];
  const float* b_hh2 = (const float*)d_in[8];
  const float* W_lin = (const float*)d_in[9];
  const float* b_lin = (const float*)d_in[10];

  int B = in_sizes[0] / (T_STEPS * F_IN);           // 8192
  dim3 grid(B / (16 * WAVES_PER_BLOCK));            // 128 blocks, 4 waves each
  lstm_wmma_kernel<<<grid, THREADS, 0, stream>>>(x, W_ih1, W_hh1, b_ih1, b_hh1,
                                                 W_ih2, W_hh2, b_ih2, b_hh2,
                                                 W_lin, b_lin, (float*)d_out);
}
